// GenGCN_80934363726534
// MI455X (gfx1250) — compile-verified, matrix operand order checked by reading, not run on hardware
//
#include <hip/hip_runtime.h>
#include <hip/hip_bf16.h>

// ---------------------------------------------------------------------------
// GCN forward for MI455X (gfx1250, wave32).
// Dense GEMMs run on bf16 WMMA (v_wmma_f32_16x16x32_bf16, f32 accum).
// Message passing = L2-resident float4 gather + f32 atomic scatter.
// Node-feature matrices (<= 51.2MB) fit in the 192MB L2, so edge traffic
// never goes to HBM; GEMM flops (~10.6 GF) are negligible on the matrix pipes.
// ---------------------------------------------------------------------------

typedef __bf16 bf16_t;
typedef __attribute__((ext_vector_type(16))) __bf16 v16bf;
typedef __attribute__((ext_vector_type(8)))  __bf16 v8bf;
typedef __attribute__((ext_vector_type(8)))  float  v8f;

__device__ __forceinline__ bf16_t f2bf(float f) {
  unsigned u = __builtin_bit_cast(unsigned, f);
  unsigned r = u + 0x7FFFu + ((u >> 16) & 1u);   // round-to-nearest-even
  unsigned short h = (unsigned short)(r >> 16);
  return __builtin_bit_cast(bf16_t, h);
}

// -------------------------------- utility ----------------------------------

__global__ void fill_f32(float* __restrict__ p, long long n, float v) {
  long long i = (long long)blockIdx.x * blockDim.x + threadIdx.x;
  long long stride = (long long)gridDim.x * blockDim.x;
  for (; i < n; i += stride) p[i] = v;
}

// WT[n*K + k] = bf16(W[k*N + n])  (transpose + downconvert, done once)
__global__ void weight_transpose_bf16(const float* __restrict__ W,
                                      bf16_t* __restrict__ WT, int K, int Nout) {
  int i = blockIdx.x * blockDim.x + threadIdx.x;
  if (i >= K * Nout) return;
  int k = i / Nout, n = i % Nout;
  WT[(size_t)n * K + k] = f2bf(W[i]);
}

// Cat[i][0:64]=x[i], Cat[i][64:96]=static[i], bf16
__global__ void build_cat(const float* __restrict__ x, const float* __restrict__ st,
                          bf16_t* __restrict__ cat, int Nn) {
  long long i = (long long)blockIdx.x * blockDim.x + threadIdx.x;
  long long total = (long long)Nn * 96;
  if (i >= total) return;
  int row = (int)(i / 96), col = (int)(i % 96);
  float v = (col < 64) ? x[(size_t)row * 64 + col] : st[(size_t)row * 32 + (col - 64)];
  cat[i] = f2bf(v);
}

// ------------------------------ graph prep ---------------------------------

__global__ void degree_count(const int* __restrict__ dst, float* __restrict__ deg, int E) {
  int i = blockIdx.x * blockDim.x + threadIdx.x;
  if (i < E) atomicAdd(&deg[dst[i]], 1.0f);
}

__global__ void compute_dis(const float* __restrict__ deg, float* __restrict__ dis, int Nn) {
  int i = blockIdx.x * blockDim.x + threadIdx.x;
  if (i < Nn) dis[i] = rsqrtf(1.0f + deg[i]);
}

__global__ void compute_norm(const int* __restrict__ src, const int* __restrict__ dst,
                             const float* __restrict__ dis, float* __restrict__ norm, int E) {
  int i = blockIdx.x * blockDim.x + threadIdx.x;
  if (i < E) norm[i] = dis[src[i]] * dis[dst[i]];
}

// ------------------------------ WMMA GEMM ----------------------------------
// C[M,F] = act( A[M,K](bf16) * BT[F,K]^T(bf16) + bias )
// One wave -> one 16x16 tile. Block = 128 threads = 4 waves covering 64 cols.
// Fragment layouts per CDNA5 ISA 7.12.2 (16-bit A 16x32; C/D f32 16x16).
__global__ __launch_bounds__(128)
void gemm_bf16_wmma(const bf16_t* __restrict__ A, const bf16_t* __restrict__ BT,
                    const float* __restrict__ bias,
                    float* __restrict__ outF, bf16_t* __restrict__ outH,
                    int M, int K, int F, int relu) {
  const int wave = threadIdx.x >> 5;
  const int lane = threadIdx.x & 31;
  const int rowBase = blockIdx.x * 16;
  const int colBase = blockIdx.y * 64 + wave * 16;
  if (rowBase >= M || colBase >= F) return;

  const int halfsel = lane >> 4;   // lane[4]: selects K-half per ISA layout
  const int l15 = lane & 15;
  const bf16_t* aptr = A  + (size_t)(rowBase + l15) * K + halfsel * 8;
  const bf16_t* bptr = BT + (size_t)(colBase + l15) * K + halfsel * 8;

  v8f acc = {};
  for (int k = 0; k < K; k += 32) {
    v8bf a0 = *(const v8bf*)(aptr + k);
    v8bf a1 = *(const v8bf*)(aptr + k + 16);
    v8bf b0 = *(const v8bf*)(bptr + k);
    v8bf b1 = *(const v8bf*)(bptr + k + 16);
    v16bf av, bv;
#pragma unroll
    for (int i = 0; i < 8; ++i) {
      av[i] = a0[i]; av[i + 8] = a1[i];
      bv[i] = b0[i]; bv[i + 8] = b1[i];
    }
    acc = __builtin_amdgcn_wmma_f32_16x16x32_bf16(
        /*neg_a=*/false, av, /*neg_b=*/false, bv,
        /*c_mod=*/(short)0, acc, /*reuse_a=*/false, /*reuse_b=*/false);
  }

  const int ocol = colBase + l15;
  const float bval = bias ? bias[ocol] : 0.0f;
#pragma unroll
  for (int r = 0; r < 8; ++r) {
    int orow = rowBase + r + 8 * halfsel;   // C/D layout: VGPR r -> M=r (+8 for hi lanes)
    float v = acc[r] + bval;
    if (relu) v = fmaxf(v, 0.0f);
    size_t idx = (size_t)orow * F + ocol;
    if (outF) outF[idx] = v;
    if (outH) outH[idx] = f2bf(v);
  }
}

// --------------------------- message passing -------------------------------
// One thread per (edge, 4-feature chunk): float4 gather + 4 f32 atomic adds.
__global__ void edge_scatter(const float* __restrict__ H, const int* __restrict__ src,
                             const int* __restrict__ dst, const float* __restrict__ norm,
                             float* __restrict__ AGG, int E, int F, int fshift) {
  const long long total = (long long)E << fshift;          // E * (F/4)
  const int fmask = (1 << fshift) - 1;
  long long i = (long long)blockIdx.x * blockDim.x + threadIdx.x;
  long long stride = (long long)gridDim.x * blockDim.x;
  for (; i < total; i += stride) {
    int e  = (int)(i >> fshift);
    int f4 = ((int)i & fmask) << 2;
    int s = src[e], d = dst[e];
    float nm = norm[e];
    const float4 h = *(const float4*)(H + (size_t)s * F + f4);
    float* o = AGG + (size_t)d * F + f4;
    atomicAdd(o + 0, h.x * nm);
    atomicAdd(o + 1, h.y * nm);
    atomicAdd(o + 2, h.z * nm);
    atomicAdd(o + 3, h.w * nm);
  }
}

// out = act(AGG + H*dis^2 + b); write bf16 next-layer act and/or f32 result
__global__ void finalize_conv(const float* __restrict__ AGG, const float* __restrict__ H,
                              const float* __restrict__ dis, const float* __restrict__ bias,
                              bf16_t* __restrict__ outBF, float* __restrict__ outF32,
                              int Nn, int F, int fshift, int relu) {
  long long i = (long long)blockIdx.x * blockDim.x + threadIdx.x;
  long long total = (long long)Nn * F;
  if (i >= total) return;
  int row = (int)(i >> fshift);
  int col = (int)i & (F - 1);
  float dv = dis[row];
  float v = AGG[i] + H[i] * (dv * dv) + bias[col];
  if (relu) v = fmaxf(v, 0.0f);
  if (outBF)  outBF[i]  = f2bf(v);
  if (outF32) outF32[i] = v;
}

// ------------------------------- launcher ----------------------------------

extern "C" void kernel_launch(void* const* d_in, const int* in_sizes, int n_in,
                              void* d_out, int out_size, void* d_ws, size_t ws_size,
                              hipStream_t stream) {
  const float* x   = (const float*)d_in[0];
  const float* st  = (const float*)d_in[1];
  const int*   src = (const int*)d_in[2];
  const int*   dst = (const int*)d_in[3];
  const float* W1  = (const float*)d_in[4];  const float* b1  = (const float*)d_in[5];
  const float* W2  = (const float*)d_in[6];  const float* b2  = (const float*)d_in[7];
  const float* Wg0 = (const float*)d_in[8];  const float* bg0 = (const float*)d_in[9];
  const float* Wg1 = (const float*)d_in[10]; const float* bg1 = (const float*)d_in[11];
  const float* Wg2 = (const float*)d_in[12]; const float* bg2 = (const float*)d_in[13];
  float* out = (float*)d_out;

  const int Nn = in_sizes[0] / 64;   // 100000 (== 6250 * 16)
  const int E  = in_sizes[2];        // 1600000
  const int CAT = 96, HID = 128, OUT = 64;

  // workspace carve-out (256B aligned slabs)
  char* base = (char*)d_ws;
  size_t off = 0;
  auto take = [&](size_t bytes) -> char* {
    char* p = base + off;
    off += (bytes + 255) & ~(size_t)255;
    return p;
  };
  bf16_t* cat  = (bf16_t*)take((size_t)Nn * CAT * 2);
  bf16_t* actA = (bf16_t*)take((size_t)Nn * HID * 2);
  bf16_t* actB = (bf16_t*)take((size_t)Nn * HID * 2);
  float*  Hbuf = (float*)take((size_t)Nn * HID * 4);
  float*  AGG  = (float*)take((size_t)Nn * HID * 4);
  float*  deg  = (float*)take((size_t)Nn * 4);
  float*  dis  = (float*)take((size_t)Nn * 4);
  float*  nrm  = (float*)take((size_t)E * 4);
  bf16_t* w1t  = (bf16_t*)take((size_t)HID * CAT * 2);
  bf16_t* w2t  = (bf16_t*)take((size_t)OUT * HID * 2);
  bf16_t* wg0t = (bf16_t*)take((size_t)HID * OUT * 2);
  bf16_t* wg1t = (bf16_t*)take((size_t)HID * HID * 2);
  bf16_t* wg2t = (bf16_t*)take((size_t)OUT * HID * 2);
  (void)ws_size;

  const int T = 256;
  auto blk = [](long long n, int t) { return (unsigned)((n + t - 1) / t); };

  // 1. weight prep (transpose + bf16) — tiny, done every call for determinism
  weight_transpose_bf16<<<blk((long long)CAT * HID, T), T, 0, stream>>>(W1,  w1t,  CAT, HID);
  weight_transpose_bf16<<<blk((long long)HID * OUT, T), T, 0, stream>>>(W2,  w2t,  HID, OUT);
  weight_transpose_bf16<<<blk((long long)OUT * HID, T), T, 0, stream>>>(Wg0, wg0t, OUT, HID);
  weight_transpose_bf16<<<blk((long long)HID * HID, T), T, 0, stream>>>(Wg1, wg1t, HID, HID);
  weight_transpose_bf16<<<blk((long long)HID * OUT, T), T, 0, stream>>>(Wg2, wg2t, HID, OUT);

  // 2. concat + bf16 convert of inputs
  build_cat<<<blk((long long)Nn * CAT, T), T, 0, stream>>>(x, st, cat, Nn);

  // 3. degrees -> dis -> per-edge norm (computed once, reused by all 3 convs)
  fill_f32<<<blk(Nn, T), T, 0, stream>>>(deg, Nn, 0.0f);
  degree_count<<<blk(E, T), T, 0, stream>>>(dst, deg, E);
  compute_dis<<<blk(Nn, T), T, 0, stream>>>(deg, dis, Nn);
  compute_norm<<<blk(E, T), T, 0, stream>>>(src, dst, dis, nrm, E);

  const dim3 gB(128);
  const int mt = Nn / 16;  // 6250 row tiles

  // 4. MLP encoder: relu(cat@W1+b1) -> relu(.@W2+b2)
  gemm_bf16_wmma<<<dim3(mt, HID / 64), gB, 0, stream>>>(cat,  w1t, b1, nullptr, actA, Nn, CAT, HID, 1);
  gemm_bf16_wmma<<<dim3(mt, OUT / 64), gB, 0, stream>>>(actA, w2t, b2, nullptr, actB, Nn, HID, OUT, 1);

  // 5. conv0: 64 -> 128, relu
  gemm_bf16_wmma<<<dim3(mt, HID / 64), gB, 0, stream>>>(actB, wg0t, nullptr, Hbuf, nullptr, Nn, OUT, HID, 0);
  fill_f32<<<blk((long long)Nn * HID, T), T, 0, stream>>>(AGG, (long long)Nn * HID, 0.0f);
  edge_scatter<<<blk((long long)E * (HID / 4), T), T, 0, stream>>>(Hbuf, src, dst, nrm, AGG, E, HID, 5);
  finalize_conv<<<blk((long long)Nn * HID, T), T, 0, stream>>>(AGG, Hbuf, dis, bg0, actA, nullptr, Nn, HID, 7, 1);

  // 6. conv1: 128 -> 128, relu
  gemm_bf16_wmma<<<dim3(mt, HID / 64), gB, 0, stream>>>(actA, wg1t, nullptr, Hbuf, nullptr, Nn, HID, HID, 0);
  fill_f32<<<blk((long long)Nn * HID, T), T, 0, stream>>>(AGG, (long long)Nn * HID, 0.0f);
  edge_scatter<<<blk((long long)E * (HID / 4), T), T, 0, stream>>>(Hbuf, src, dst, nrm, AGG, E, HID, 5);
  finalize_conv<<<blk((long long)Nn * HID, T), T, 0, stream>>>(AGG, Hbuf, dis, bg1, actB, nullptr, Nn, HID, 7, 1);

  // 7. conv2: 128 -> 64, no relu, f32 output
  gemm_bf16_wmma<<<dim3(mt, OUT / 64), gB, 0, stream>>>(actB, wg2t, nullptr, Hbuf, nullptr, Nn, HID, OUT, 0);
  fill_f32<<<blk((long long)Nn * OUT, T), T, 0, stream>>>(AGG, (long long)Nn * OUT, 0.0f);
  edge_scatter<<<blk((long long)E * (OUT / 4), T), T, 0, stream>>>(Hbuf, src, dst, nrm, AGG, E, OUT, 4);
  finalize_conv<<<blk((long long)Nn * OUT, T), T, 0, stream>>>(AGG, Hbuf, dis, bg2, nullptr, out, Nn, OUT, 6, 0);
}
